// HaloAttention_52475910422609
// MI455X (gfx1250) — matrix operand order repaired and checked by reading, not compiled, not run on hardware
//
#include <hip/hip_runtime.h>
#include <hip/hip_fp16.h>

// ---------------------------------------------------------------------------
// HaloAttention fwd for MI455X (gfx1250, wave32, WMMA f32_16x16x32_f16)
// ---------------------------------------------------------------------------

typedef _Float16 h8   __attribute__((ext_vector_type(8)));
typedef _Float16 v16h __attribute__((ext_vector_type(16)));
typedef float    v8f  __attribute__((ext_vector_type(8)));

union VH { v16h v; h8 h[2]; };

// A fragment (16x32 f16): lane = row (lane&15); chunks at 8*hf and 16+8*hf (ISA 7.12.2)
__device__ __forceinline__ v16h load_a_frag(const _Float16* base, int ld, int row, int hf) {
    const _Float16* p = base + row * ld;
    VH u;
    u.h[0] = *(const h8*)(p + 8 * hf);
    u.h[1] = *(const h8*)(p + 16 + 8 * hf);
    return u.v;
}
// B fragment (32x16 f16): lane = col (lane&15); chunks at 8*hf and 16+8*hf
__device__ __forceinline__ v16h load_b_frag(const _Float16* base, int ld, int col, int hf) {
    const _Float16* p = base + col * ld;
    VH u;
    u.h[0] = *(const h8*)(p + 8 * hf);
    u.h[1] = *(const h8*)(p + 16 + 8 * hf);
    return u.v;
}
__device__ __forceinline__ v8f wmma16(v16h a, v16h b, v8f c) {
    return __builtin_amdgcn_wmma_f32_16x16x32_f16(false, a, false, b, (short)0, c, false, false);
}

#define NEG_MAX (-3.402823466e38f)

// ---------------------------------------------------------------------------
// Kernel 0: weight prep (transpose to [N][K] f16, fold q-scale, pad rel tables)
// ---------------------------------------------------------------------------
__global__ void halo_prep(const float* __restrict__ wq, const float* __restrict__ wkv,
                          const float* __restrict__ wout,
                          const float* __restrict__ relh, const float* __restrict__ relw,
                          _Float16* __restrict__ wt,      // [1536][256]  (wq_t || wkv_t)
                          _Float16* __restrict__ woutt,   // [256][512]
                          _Float16* __restrict__ relw_h,  // [32][64] row31=0
                          _Float16* __restrict__ relh_h) {
    int idx = blockIdx.x * 256 + threadIdx.x;           // grid covers 262144
    if (idx < 512 * 256) {                              // wq_t[n][c], scale = 1/sqrt(64)
        int n = idx >> 8, c = idx & 255;
        wt[idx] = (_Float16)(wq[c * 512 + n] * 0.125f);
    }
    if (idx < 1024 * 256) {                             // wkv_t[n][c]
        int n = idx >> 8, c = idx & 255;
        wt[512 * 256 + idx] = (_Float16)(wkv[c * 1024 + n]);
    }
    if (idx < 256 * 512) {                              // wout_t[n][k]
        int n = idx >> 9, k = idx & 511;
        woutt[idx] = (_Float16)(wout[k * 256 + n]);
    }
    if (idx < 2048) {
        int r = idx >> 6, d = idx & 63;
        relw_h[idx] = (_Float16)(r < 31 ? relw[r * 64 + d] : 0.0f);
        relh_h[idx] = (_Float16)(r < 31 ? relh[r * 64 + d] : 0.0f);
    }
}

// ---------------------------------------------------------------------------
// Kernel 1: fused QKV projection.  M=32768 positions, K=256, N=1536.
//  Q  -> qws[b][pos][head*64+d] (pre-scaled)
//  K  -> kws[b][head][pos][d]
//  V  -> vws[b][head][d][pos]   (transposed for attn@V B-fragments)
// ---------------------------------------------------------------------------
__global__ void halo_proj_qkv(const float* __restrict__ x, const _Float16* __restrict__ wt,
                              _Float16* __restrict__ qws, _Float16* __restrict__ kws,
                              _Float16* __restrict__ vws) {
    __shared__ _Float16 sA[64 * 40];                    // 64 positions x 32 chans, pad 40
    const int tid = threadIdx.x, lane = tid & 31, wv = tid >> 5;
    const int ln = lane & 15, hf = lane >> 4;
    const int Mbase = blockIdx.x * 64, Nbase = blockIdx.y * 64;
    const int b = Mbase >> 12, posbase = Mbase & 4095;

    v8f acc[4] = {};
    for (int kc = 0; kc < 8; ++kc) {
        const int kb = kc * 32;
        for (int it = 0; it < 16; ++it) {               // stage A tile (f32 -> f16)
            int idx = tid + 128 * it;                   // 2048 elements
            int c = idx >> 6, p = idx & 63;
            sA[p * 40 + c] = (_Float16)(x[(b * 256 + kb + c) * 4096 + posbase + p]);
        }
        __syncthreads();
        v16h a = load_a_frag(sA, 40, 16 * wv + ln, hf);
#pragma unroll
        for (int nt = 0; nt < 4; ++nt) {
            v16h bf = load_b_frag(wt + kb, 256, Nbase + nt * 16 + ln, hf);
            acc[nt] = wmma16(a, bf, acc[nt]);
        }
        __syncthreads();
    }
#pragma unroll
    for (int nt = 0; nt < 4; ++nt) {
        int n = Nbase + nt * 16 + ln;
#pragma unroll
        for (int r = 0; r < 8; ++r) {
            int pos = posbase + 16 * wv + r + 8 * hf;
            _Float16 v = (_Float16)acc[nt][r];
            if (n < 512) {
                qws[(b * 4096 + pos) * 512 + n] = v;
            } else if (n < 1024) {
                int nk = n - 512, head = nk >> 6, d = nk & 63;
                kws[((b * 8 + head) * 4096 + pos) * 64 + d] = v;
            } else {
                int nv = n - 1024, head = nv >> 6, d = nv & 63;
                vws[((b * 8 + head) * 64 + d) * 4096 + pos] = v;
            }
        }
    }
}

// ---------------------------------------------------------------------------
// Kernel 2: attention per (batch, block, head).  4 waves x wave32.
// ---------------------------------------------------------------------------
#define LDQ   72
#define LDK   72
#define LDVT  264
#define LDS16 264
#define LDSF  257
#define LDP   66
#define SMEM_ATTN ((64*LDQ + 256*LDK + 64*LDVT + 64*LDS16)*2 + (64*LDSF + 64*LDP)*4)

__global__ void halo_attn(const _Float16* __restrict__ qws, const _Float16* __restrict__ kws,
                          const _Float16* __restrict__ vws,
                          const _Float16* __restrict__ relw_h, const _Float16* __restrict__ relh_h,
                          _Float16* __restrict__ out1) {
    extern __shared__ char smem[];
    _Float16* sQ   = (_Float16*)smem;                   // [64][LDQ]
    _Float16* sK   = sQ + 64 * LDQ;                     // [256][LDK]
    _Float16* sVt  = sK + 256 * LDK;                    // [64][LDVT]  (V transposed: d x j)
    _Float16* sS16 = sVt + 64 * LDVT;                   // [64][LDS16] softmaxed attn (f16)
    float*    sSf  = (float*)(sS16 + 64 * LDS16);       // [64][LDSF]  raw logits (f32)
    float*    sP   = sSf + 64 * LDSF;                   // [64][LDP]   rel logits (w:0..31, h:32..63)

    const int tid = threadIdx.x, lane = tid & 31, wv = tid >> 5;
    const int ln = lane & 15, hf = lane >> 4;
    const int id = blockIdx.x;
    const int head = id & 7, blk = (id >> 3) & 63, b = id >> 9;
    const int by = blk >> 3, bx = blk & 7;

    const _Float16* qbase = qws + (size_t)(b * 4096) * 512 + head * 64;
    const _Float16* kbase = kws + (size_t)((b * 8 + head) * 4096) * 64;
    const _Float16* vbase = vws + (size_t)((b * 8 + head) * 64) * 4096;

    // ---- stage Q (64x64), K window (256x64, zero halo), V^T (64x256, zero halo)
    for (int it = 0; it < 4; ++it) {
        int idx = tid + 128 * it;                       // 512 chunks of 8 halfs
        int row = idx >> 3, c8 = idx & 7;
        int py = by * 8 + (row >> 3), px = bx * 8 + (row & 7);
        *(h8*)(sQ + row * LDQ + c8 * 8) = *(const h8*)(qbase + (py * 64 + px) * 512 + c8 * 8);
    }
    for (int it = 0; it < 16; ++it) {
        int idx = tid + 128 * it;                       // 2048 chunks
        int j = idx >> 3, c8 = idx & 7;
        int ky = j >> 4, kx = j & 15;
        int py = by * 8 - 4 + ky, px = bx * 8 - 4 + kx;
        h8 val = {};
        if (py >= 0 && py < 64 && px >= 0 && px < 64)
            val = *(const h8*)(kbase + (py * 64 + px) * 64 + c8 * 8);
        *(h8*)(sK + j * LDK + c8 * 8) = val;
    }
    for (int it = 0; it < 16; ++it) {
        int idx = tid + 128 * it;                       // d*32 + cc
        int d = idx >> 5, cc = idx & 31;
        int ky = cc >> 1, kx0 = (cc & 1) * 8;
        int py = by * 8 - 4 + ky, s = bx * 8 - 4 + kx0;
        h8 val = {};
        if (py >= 0 && py < 64) {
            if (s >= 0 && s + 7 < 64) {
                val = *(const h8*)(vbase + d * 4096 + py * 64 + s);
            } else {
                for (int e = 0; e < 8; ++e) {
                    int px = s + e;
                    if (px >= 0 && px < 64) val[e] = vbase[d * 4096 + py * 64 + px];
                }
            }
        }
        *(h8*)(sVt + d * LDVT + cc * 8) = val;
    }
    __syncthreads();

    // ---- rel-pos: P[:,0:32] = Q @ relW^T ; P[:,32:64] = Q @ relH^T   (WMMA)
    v16h a0 = load_a_frag(sQ, LDQ, 16 * wv + ln, hf);
    v16h a1 = load_a_frag(sQ + 32, LDQ, 16 * wv + ln, hf);
#pragma unroll
    for (int part = 0; part < 2; ++part) {
        const _Float16* R = part ? relh_h : relw_h;
#pragma unroll
        for (int nt = 0; nt < 2; ++nt) {
            v8f acc = {};
            acc = wmma16(a0, load_b_frag(R, 64, nt * 16 + ln, hf), acc);
            acc = wmma16(a1, load_b_frag(R + 32, 64, nt * 16 + ln, hf), acc);
#pragma unroll
            for (int r = 0; r < 8; ++r) {
                int i = 16 * wv + r + 8 * hf;
                sP[i * LDP + part * 32 + nt * 16 + ln] = acc[r];
            }
        }
    }
    __syncthreads();

    // ---- S = Q @ K^T + rel, then faithful mask (in-image keys -> -FLT_MAX)
    for (int jt = 0; jt < 16; ++jt) {
        v8f acc = {};
        acc = wmma16(a0, load_b_frag(sK, LDK, jt * 16 + ln, hf), acc);
        acc = wmma16(a1, load_b_frag(sK + 32, LDK, jt * 16 + ln, hf), acc);
        const int kx = ln, ky = jt;
        const int py = by * 8 - 4 + ky, px = bx * 8 - 4 + kx;
        const bool inside = (py >= 0 && py < 64 && px >= 0 && px < 64);
#pragma unroll
        for (int r = 0; r < 8; ++r) {
            int i = 16 * wv + r + 8 * hf;
            int qy = i >> 3, qx = i & 7;
            float v = acc[r] + sP[i * LDP + (kx - qx + 15)]
                             + sP[i * LDP + 32 + (ky - qy + 15)];
            if (inside) v = NEG_MAX;                    // faithful to reference mask
            sSf[i * LDSF + jt * 16 + kx] = v;
        }
    }
    __syncthreads();

    // ---- stable softmax over J=256 (2 lanes per row, combine via shfl_xor 16)
    {
        int row = 16 * wv + ln;
        const float* rp = sSf + row * LDSF + hf * 128;
        float mx = NEG_MAX;
        for (int c = 0; c < 128; ++c) mx = fmaxf(mx, rp[c]);
        mx = fmaxf(mx, __shfl_xor(mx, 16, 32));
        float sum = 0.0f;
        for (int c = 0; c < 128; ++c) sum += __expf(rp[c] - mx);
        sum += __shfl_xor(sum, 16, 32);
        float inv = 1.0f / sum;
        _Float16* op = sS16 + row * LDS16 + hf * 128;
        for (int c = 0; c < 128; ++c) op[c] = (_Float16)(__expf(rp[c] - mx) * inv);
    }
    __syncthreads();

    // ---- O = attn @ V  (B-fragments contiguous thanks to V^T layout)
    {
        v16h afr[8];
#pragma unroll
        for (int kc = 0; kc < 8; ++kc)
            afr[kc] = load_a_frag(sS16 + kc * 32, LDS16, 16 * wv + ln, hf);
        const int rowbase = (b * 64 + blk) * 64 + 16 * wv;
#pragma unroll
        for (int dt = 0; dt < 4; ++dt) {
            v8f acc = {};
#pragma unroll
            for (int kc = 0; kc < 8; ++kc)
                acc = wmma16(afr[kc], load_b_frag(sVt + kc * 32, LDVT, dt * 16 + ln, hf), acc);
#pragma unroll
            for (int r = 0; r < 8; ++r) {
                int i = rowbase + r + 8 * hf;
                out1[(size_t)i * 512 + head * 64 + dt * 16 + ln] = (_Float16)acc[r];
            }
        }
    }
}

// ---------------------------------------------------------------------------
// Kernel 3: out projection [32768x512]@[512x256] + bias, scatter to NCHW f32
// ---------------------------------------------------------------------------
__global__ void halo_proj_out(const _Float16* __restrict__ out1, const _Float16* __restrict__ woutt,
                              const float* __restrict__ bout, float* __restrict__ out) {
    const int tid = threadIdx.x, lane = tid & 31, wv = tid >> 5;
    const int ln = lane & 15, hf = lane >> 4;
    const int Mbase = blockIdx.x * 64, Nbase = blockIdx.y * 64;

    v8f acc[4] = {};
    const _Float16* ap = out1 + (size_t)(Mbase + 16 * wv + ln) * 512;
    for (int kc = 0; kc < 16; ++kc) {
        v16h a = load_a_frag(ap + kc * 32, 0, 0, hf);   // direct-from-global A fragment
#pragma unroll
        for (int nt = 0; nt < 4; ++nt)
            acc[nt] = wmma16(a, load_b_frag(woutt + kc * 32, 512, Nbase + nt * 16 + ln, hf), acc[nt]);
    }
#pragma unroll
    for (int nt = 0; nt < 4; ++nt) {
        int c = Nbase + nt * 16 + ln;
        float bias = bout[c];
#pragma unroll
        for (int r = 0; r < 8; ++r) {
            int rowg = Mbase + 16 * wv + r + 8 * hf;
            int b = rowg >> 12, rem = rowg & 4095;
            int blk = rem >> 6, i = rem & 63;
            int by = blk >> 3, bx = blk & 7;
            int py = by * 8 + (i >> 3), px = bx * 8 + (i & 7);
            out[((size_t)(b * 256 + c) << 12) + (py << 6) + px] = acc[nt][r] + bias;
        }
    }
}

// ---------------------------------------------------------------------------
// workspace layout (bytes); total ~129 MB
// ---------------------------------------------------------------------------
static const size_t OFF_WT   = 0;                       // 1536*256 f16
static const size_t OFF_WOUT = 786432;                  // 256*512  f16
static const size_t OFF_RELW = 1048576;                 // 32*64    f16
static const size_t OFF_RELH = 1052672;                 // 32*64    f16
static const size_t OFF_Q    = 1056768;                 // 8*4096*512 f16
static const size_t OFF_K    = OFF_Q + 33554432;        // 8*8*4096*64 f16
static const size_t OFF_V    = OFF_K + 33554432;        // transposed
static const size_t OFF_O1   = OFF_V + 33554432;        // 32768*512 f16

extern "C" void kernel_launch(void* const* d_in, const int* in_sizes, int n_in,
                              void* d_out, int out_size, void* d_ws, size_t ws_size,
                              hipStream_t stream) {
    const float* x    = (const float*)d_in[0];
    const float* wq   = (const float*)d_in[1];
    const float* wkv  = (const float*)d_in[2];
    const float* wout = (const float*)d_in[3];
    const float* bout = (const float*)d_in[4];
    const float* relh = (const float*)d_in[5];
    const float* relw = (const float*)d_in[6];

    char* ws = (char*)d_ws;
    _Float16* wt     = (_Float16*)(ws + OFF_WT);
    _Float16* woutt  = (_Float16*)(ws + OFF_WOUT);
    _Float16* relw_h = (_Float16*)(ws + OFF_RELW);
    _Float16* relh_h = (_Float16*)(ws + OFF_RELH);
    _Float16* qws    = (_Float16*)(ws + OFF_Q);
    _Float16* kws    = (_Float16*)(ws + OFF_K);
    _Float16* vws    = (_Float16*)(ws + OFF_V);
    _Float16* o1     = (_Float16*)(ws + OFF_O1);

    (void)in_sizes; (void)n_in; (void)out_size; (void)ws_size;

    // 192 KB dynamic LDS for the attention kernel (CDNA5: up to 320 KB / WG)
    hipFuncSetAttribute((const void*)halo_attn,
                        hipFuncAttributeMaxDynamicSharedMemorySize, SMEM_ATTN);

    halo_prep<<<1024, 256, 0, stream>>>(wq, wkv, wout, relh, relw,
                                        wt, woutt, relw_h, relh_h);
    halo_proj_qkv<<<dim3(512, 24), 128, 0, stream>>>(x, wt, qws, kws, vws);
    halo_attn<<<4096, 128, SMEM_ATTN, stream>>>(qws, kws, vws, relw_h, relh_h, o1);
    halo_proj_out<<<dim3(512, 4), 128, 0, stream>>>(o1, woutt, bout, (float*)d_out);
}
